// P6DenseMojoFlashAttention_44470091383152
// MI455X (gfx1250) — matrix operand order repaired and checked by reading, not compiled, not run on hardware
//
#include <hip/hip_runtime.h>
#include <hip/hip_bf16.h>

// Problem constants (match reference: H=16, T=4096, D=64, NSEQ=8)
#define HH 16
#define TT 4096
#define DD 64
#define BQ 256   // q rows per workgroup (8 waves x 2 tiles x 16 rows)
#define KB 32    // keys per inner block (one WMMA K for PV)

#define KSTR 72  // K tile row stride in bf16 (16B-aligned rows, bank-spread)
#define VSTR 40  // V^T tile row stride in bf16 (80B rows: 16B-aligned, bank-spread)
#define PSTR 40  // P buffer row stride in bf16

typedef __attribute__((ext_vector_type(16))) __bf16 v16bf;
typedef __attribute__((ext_vector_type(8)))  __bf16 v8bf;
typedef __attribute__((ext_vector_type(8)))  float  v8f;

#define CAT16(lo, hi) __builtin_shufflevector(lo, hi, 0, 1, 2, 3, 4, 5, 6, 7, \
                                              8, 9, 10, 11, 12, 13, 14, 15)

__device__ __forceinline__ float red_max16(float v) {
#pragma unroll
  for (int m = 1; m < 16; m <<= 1) v = fmaxf(v, __shfl_xor(v, m, 32));
  return v;
}

__global__ __launch_bounds__(256)
void fa_varlen_bf16_wmma(const float* __restrict__ qg,
                         const float* __restrict__ kg,
                         const float* __restrict__ vg,
                         const int*   __restrict__ cu,
                         int nseq,
                         float* __restrict__ out)
{
  // Double-buffered bf16 tiles: K row-major [key][d], V transposed [d][key],
  // plus per-wave, per-qtile bf16 P transpose buffers [row][key].
  __shared__ __bf16 KtS[2][KB * KSTR];
  __shared__ __bf16 VtS[2][DD * VSTR];
  __shared__ __bf16 PbS[8][2][16 * PSTR];

  const int h    = blockIdx.y;
  const int q0   = blockIdx.x * BQ;
  const int wid  = threadIdx.x >> 5;
  const int lane = threadIdx.x & 31;
  const int hf   = lane >> 4;    // half-wave select (wave32)
  const int n16  = lane & 15;

  // Segment bounds from cu_seqlens (uniform across the block).
  int seg_start = 0, seg_end = TT;
  for (int s = 0; s < nseq; ++s) {
    int a = cu[s], b = cu[s + 1];
    if (q0 >= a && q0 < b) { seg_start = a; seg_end = b; }
  }

  const float* qh = qg + (size_t)h * TT * DD;
  const float* kh = kg + (size_t)h * TT * DD;
  const float* vh = vg + (size_t)h * TT * DD;

  // ---- Load this wave's two 16-row Q tiles straight into WMMA A layout (bf16).
  // A element j (=8g+r) of lane l holds Q[row = l&15][d = 32w + 16*hf + 8g + r].
  int qr0[2];
  qr0[0] = q0 + wid * 16;
  qr0[1] = q0 + 128 + wid * 16;
  v16bf qa[2][2];   // [qtile][w]
#pragma unroll
  for (int u = 0; u < 2; ++u) {
#pragma unroll
    for (int w = 0; w < 2; ++w) {
#pragma unroll
      for (int g = 0; g < 2; ++g) {
        const float* src = qh + (size_t)(qr0[u] + n16) * DD + 32 * w + 16 * hf + 8 * g;
        float tmp[8];
        *(float4*)(tmp + 0) = *(const float4*)(src + 0);
        *(float4*)(tmp + 4) = *(const float4*)(src + 4);
#pragma unroll
        for (int r = 0; r < 8; ++r) qa[u][w][8 * g + r] = (__bf16)tmp[r];
      }
    }
  }

  // Constant all-ones B tile: row-sum of P via the tensor core (P @ 1).
  v16bf ones_b;
#pragma unroll
  for (int j = 0; j < 16; ++j) ones_b[j] = (__bf16)1.0f;

  // Staging assignment: each of 256 threads owns 8 contiguous d of one key.
  const int sf   = threadIdx.x * 8;   // 0..2047
  const int skey = sf >> 6;           // 0..31
  const int sd   = sf & 63;           // multiple of 8

  float4 kA, kB4, vA, vB4;            // in-flight global data (register stage)
  auto load_regs = [&](int blk) {
    int gk = blk + skey; if (gk > TT - 1) gk = TT - 1;
    const float* kp = kh + (size_t)gk * DD + sd;
    const float* vp = vh + (size_t)gk * DD + sd;
    kA  = *(const float4*)kp;  kB4 = *(const float4*)(kp + 4);
    vA  = *(const float4*)vp;  vB4 = *(const float4*)(vp + 4);
  };
  auto store_lds = [&](int b) {
    float kt[8], vt[8];
    *(float4*)(kt + 0) = kA;  *(float4*)(kt + 4) = kB4;
    *(float4*)(vt + 0) = vA;  *(float4*)(vt + 4) = vB4;
    v8bf kb16;
#pragma unroll
    for (int i = 0; i < 8; ++i) kb16[i] = (__bf16)kt[i];
    *(v8bf*)&KtS[b][skey * KSTR + sd] = kb16;        // one ds_store_b128
#pragma unroll
    for (int i = 0; i < 8; ++i)                       // transposed scatter
      VtS[b][(sd + i) * VSTR + skey] = (__bf16)vt[i];
  };

  // Flash-attention running state per qtile. Per lane: rows r + 8*hf.
  float ms[2][8];
  v8f o[2][4];
  v8f osum[2];
#pragma unroll
  for (int u = 0; u < 2; ++u) {
#pragma unroll
    for (int r = 0; r < 8; ++r) ms[u][r] = -__builtin_inff();
#pragma unroll
    for (int t = 0; t < 4; ++t) o[u][t] = {};
    osum[u] = {};
  }

  // scores scaled by sqrt(D) * log2(e) so softmax uses native v_exp_f32.
  const float scale_l2e = 8.0f * 1.44269504088896f;

  // Prologue: fill buffer 0.
  load_regs(seg_start);
  store_lds(0);
  int cur = 0;

  for (int kb = seg_start; kb < seg_end; kb += KB) {
    const int nb = kb + KB;
    const bool have_next = nb < seg_end;
    if (have_next) load_regs(nb);        // next block's global loads in flight
    if (nb + KB < seg_end) {             // prefetch block after next
      int pr = nb + KB + (threadIdx.x >> 3); if (pr > TT - 1) pr = TT - 1;
      __builtin_prefetch(kh + (size_t)pr * DD, 0, 1);
      __builtin_prefetch(vh + (size_t)pr * DD, 0, 1);
    }
    __syncthreads();                     // buf[cur] ready; prior reads retired

    // ---- S = (Q K^T) : each K B-tile loaded once, consumed by both q tiles.
    const bool val0 = (kb + n16)      < seg_end;
    const bool val1 = (kb + 16 + n16) < seg_end;
    v8f cacc[2][2];                      // [qtile][key-tile]
#pragma unroll
    for (int u = 0; u < 2; ++u) { cacc[u][0] = {}; cacc[u][1] = {}; }
#pragma unroll
    for (int t = 0; t < 2; ++t) {
      const int key = t * 16 + n16;      // B operand column n = lane&15
#pragma unroll
      for (int w = 0; w < 2; ++w) {
        // B element j -> d = 32w + 16*hf + j : 16 contiguous bf16 in LDS.
        const __bf16* kr = &KtS[cur][key * KSTR + 32 * w + 16 * hf];
        v8bf lo = *(const v8bf*)(kr);
        v8bf hi = *(const v8bf*)(kr + 8);
        v16bf bk = CAT16(lo, hi);
#pragma unroll
        for (int u = 0; u < 2; ++u)      // two independent WMMA chains
          cacc[u][t] = __builtin_amdgcn_wmma_f32_16x16x32_bf16(
              false, qa[u][w], false, bk, (short)0, cacc[u][t], false, false);
      }
    }

    // ---- Online softmax per qtile/row. Row max via 16-lane butterfly; row
    // sums ride the tensor core (osum vs ones_b). Invalid keys carry -inf
    // scores; raw v_exp_f32(-inf) == 0, so no branches.
    v16bf pa[2];
#pragma unroll
    for (int u = 0; u < 2; ++u) {
      __bf16* Pw = PbS[wid][u];
#pragma unroll
      for (int r = 0; r < 8; ++r) {
        float s0 = val0 ? cacc[u][0][r] * scale_l2e : -__builtin_inff();
        float s1 = val1 ? cacc[u][1][r] * scale_l2e : -__builtin_inff();
        float mnew = fmaxf(ms[u][r], red_max16(fmaxf(s0, s1)));
        float p0 = __builtin_amdgcn_exp2f(s0 - mnew);
        float p1 = __builtin_amdgcn_exp2f(s1 - mnew);
        float corr = __builtin_amdgcn_exp2f(ms[u][r] - mnew);
        ms[u][r] = mnew;
        const int row = r + 8 * hf;      // C-layout row for this lane half
        Pw[row * PSTR + n16]      = (__bf16)p0;
        Pw[row * PSTR + 16 + n16] = (__bf16)p1;
        osum[u][r] *= corr;
#pragma unroll
        for (int t = 0; t < 4; ++t) o[u][t][r] *= corr;
      }
      // Reload P in WMMA A layout: element j -> key 16*hf + j, keys
      // contiguous -> two ds_load_b128.
      const __bf16* pr = &Pw[n16 * PSTR + 16 * hf];
      v8bf plo = *(const v8bf*)(pr);        // keys 16*hf + 0..7
      v8bf phi = *(const v8bf*)(pr + 8);    // keys 16*hf + 8..15
      pa[u] = CAT16(plo, phi);
    }

    // Row sums: osum += P @ ones (replaces ds_bpermute sum reductions).
#pragma unroll
    for (int u = 0; u < 2; ++u)
      osum[u] = __builtin_amdgcn_wmma_f32_16x16x32_bf16(
          false, pa[u], false, ones_b, (short)0, osum[u], false, false);

    // ---- O += P V : each V B-tile loaded once, consumed by both q tiles.
#pragma unroll
    for (int t = 0; t < 4; ++t) {
      // V^T layout: B element j -> key = 16*hf + j, contiguous bf16.
      const __bf16* vr = &VtS[cur][(t * 16 + n16) * VSTR + 16 * hf];
      v8bf lo = *(const v8bf*)(vr);
      v8bf hi = *(const v8bf*)(vr + 8);
      v16bf bv = CAT16(lo, hi);
#pragma unroll
      for (int u = 0; u < 2; ++u)
        o[u][t] = __builtin_amdgcn_wmma_f32_16x16x32_bf16(
            false, pa[u], false, bv, (short)0, o[u][t], false, false);
    }

    if (have_next) store_lds(cur ^ 1);   // fill other buffer for next iter
    cur ^= 1;
  }

  // ---- Epilogue: normalize (v_rcp_f32) and store [T, H, D].
#pragma unroll
  for (int u = 0; u < 2; ++u) {
#pragma unroll
    for (int r = 0; r < 8; ++r) {
      const float inv = __builtin_amdgcn_rcpf(osum[u][r]);
      const int row = qr0[u] + r + 8 * hf;
#pragma unroll
      for (int t = 0; t < 4; ++t)
        out[(size_t)row * (HH * DD) + h * DD + t * 16 + n16] = o[u][t][r] * inv;
    }
  }
}

extern "C" void kernel_launch(void* const* d_in, const int* in_sizes, int n_in,
                              void* d_out, int out_size, void* d_ws, size_t ws_size,
                              hipStream_t stream) {
  const float* q  = (const float*)d_in[0];
  const float* k  = (const float*)d_in[1];
  const float* v  = (const float*)d_in[2];
  const int*   cu = (const int*)d_in[3];
  const int nseq  = in_sizes[3] - 1;
  float* out = (float*)d_out;
  (void)n_in; (void)out_size; (void)d_ws; (void)ws_size;

  dim3 grid(TT / BQ, HH);   // 16 q-blocks x 16 heads
  fa_varlen_bf16_wmma<<<grid, 256, 0, stream>>>(q, k, v, cu, nseq, out);
}
